// RPN_Network_81922206204093
// MI455X (gfx1250) — matrix operand order repaired and checked by reading, not compiled, not run on hardware
//
#include <hip/hip_runtime.h>
#include <hip/hip_bf16.h>
#include <stdint.h>

// ---------- types for WMMA operands ----------
typedef __bf16 v16bf __attribute__((ext_vector_type(16)));
typedef float  v8f   __attribute__((ext_vector_type(8)));
typedef unsigned int u32;
typedef u32 u32x4 __attribute__((ext_vector_type(4)));
typedef float f32x4 __attribute__((ext_vector_type(4)));

union ABReg { u32x4 q[2]; v16bf v; };

// ---------- problem constants ----------
#define IMG_H   100
#define IMG_W   100
#define NPIX    10000
#define IN_CH   512
#define MID     512
#define KDIM    4608          // 512 * 9
#define KB_CONV 144           // 4608 / 32
#define KB_HEAD 16            // 512 / 32
#define PADW    102
#define NANCH   9
#define NPROP   90000
#define PRE_NMS 12000
#define POST_NMS 2000
#define NBINS   4096
#define STRIDE_F 16.0f
#define CLIP_MAX 1600.0f
#define MIN_SIZE 16.0f
#define NMS_THR  0.7f
#define BN_EPS   1e-5f

// output offsets (floats)
#define OUT_CONV1 0
#define OUT_ROI   5120000
#define OUT_BND   5128000
#define OUT_CLS   5488000

// ---------- workspace layout (bytes, all 16B-aligned) ----------
static constexpr size_t XP_OFF   = 0;                                   // padded HWC bf16 input
static constexpr size_t XP_SZ    = (size_t)PADW*PADW*IN_CH*2;           // 10,653,696
static constexpr size_t WC_OFF   = XP_OFF + XP_SZ;
static constexpr size_t WC_SZ    = (size_t)KB_CONV*32*32*8*4;           // 4,718,592
static constexpr size_t CP_OFF   = WC_OFF + WC_SZ;                      // conv pre-BN [512][10000] f32
static constexpr size_t CP_SZ    = (size_t)MID*NPIX*4;
static constexpr size_t MV_OFF   = CP_OFF + CP_SZ;                      // mean[512], var[512]
static constexpr size_t MV_SZ    = 1024*4;
static constexpr size_t C1H_OFF  = MV_OFF + MV_SZ;                      // conv1 HWC bf16
static constexpr size_t C1H_SZ   = (size_t)NPIX*MID*2;
static constexpr size_t WH_OFF   = C1H_OFF + C1H_SZ;                    // packed head weights
static constexpr size_t WH_SZ    = (size_t)KB_HEAD*4*32*8*4;
static constexpr size_t G_OFF    = WH_OFF + WH_SZ;                      // head GEMM out [10000][64] f32
static constexpr size_t G_SZ     = (size_t)NPIX*64*4;
static constexpr size_t ROIS_OFF = G_OFF + G_SZ;                        // rois [90000][4]
static constexpr size_t ROIS_SZ  = (size_t)NPROP*4*4;
static constexpr size_t SCORE_OFF= ROIS_OFF + ROIS_SZ;                  // score [90000]
static constexpr size_t SCORE_SZ = (size_t)NPROP*4;
static constexpr size_t HIST_OFF = SCORE_OFF + SCORE_SZ;
static constexpr size_t HIST_SZ  = (size_t)NBINS*4;
static constexpr size_t CTRS_OFF = HIST_OFF + HIST_SZ;                  // counters/threshold
static constexpr size_t CTRS_SZ  = 64;
static constexpr size_t SELB_OFF = CTRS_OFF + CTRS_SZ;                  // selected boxes [12000][4]
static constexpr size_t SELB_SZ  = (size_t)PRE_NMS*4*4;
static constexpr size_t SELS_OFF = SELB_OFF + SELB_SZ;                  // selected scores [12000]

// ---------- helpers ----------
__device__ __forceinline__ unsigned short f2bf(float f) {
    u32 u = __float_as_uint(f);
    u32 r = u + 0x7FFFu + ((u >> 16) & 1u);   // round-to-nearest-even
    return (unsigned short)(r >> 16);
}
__device__ __forceinline__ int score_bin(float s) {
    u32 u = __float_as_uint(s);
    u = (u & 0x80000000u) ? ~u : (u | 0x80000000u);   // monotone total order
    return (int)(u >> 20);
}

// ---------- kernels ----------
__global__ void k_init(int* hist_ctrs) {
    int i = blockIdx.x * blockDim.x + threadIdx.x;
    if (i < NBINS + 16) hist_ctrs[i] = 0;
}

// fp32 NCHW -> padded (102x102) HWC bf16
__global__ void k_pack_x(const float* __restrict__ x, unsigned short* __restrict__ xp) {
    int idx = blockIdx.x * blockDim.x + threadIdx.x;
    if (idx >= PADW * PADW * IN_CH) return;
    int c = idx % IN_CH, t = idx / IN_CH;
    int xx = t % PADW, yy = t / PADW;
    float v = 0.0f;
    if (yy >= 1 && yy <= IMG_H && xx >= 1 && xx <= IMG_W)
        v = x[c * NPIX + (yy - 1) * IMG_W + (xx - 1)];
    xp[idx] = f2bf(v);
}

// conv weights [512][512][3][3] -> B-operand packed [kb][ntile][lane][8 dwords]
// K ordering: K = (ky*3+kx)*512 + cin ; B layout: K = g*16 + 2v + h, N = lane%16
__global__ void k_pack_wconv(const float* __restrict__ w, u32* __restrict__ wp) {
    int idx = blockIdx.x * blockDim.x + threadIdx.x;
    if (idx >= KB_CONV * 32 * 32 * 8) return;
    int d  = idx & 7;
    int l  = (idx >> 3) & 31;
    int nt = (idx >> 8) & 31;
    int kb = idx >> 13;
    int n = nt * 16 + (l & 15);
    int g = l >> 4;
    int K = kb * 32 + g * 16 + 2 * d;
    int seg0 = K >> 9,       cin0 = K & 511;
    int seg1 = (K + 1) >> 9, cin1 = (K + 1) & 511;
    unsigned short lo = f2bf(w[(size_t)n * KDIM + cin0 * 9 + seg0]);
    unsigned short hi = f2bf(w[(size_t)n * KDIM + cin1 * 9 + seg1]);
    wp[idx] = (u32)lo | ((u32)hi << 16);
}

// implicit-GEMM conv: D[16x16] += A(16x32 bf16) x B(32x16 bf16), 144 K-blocks
__global__ void __launch_bounds__(128) k_conv_wmma(const unsigned short* __restrict__ xp,
                                                   const u32* __restrict__ wp,
                                                   float* __restrict__ conv_pre) {
    const int lane = threadIdx.x & 31;
    const int wv   = threadIdx.x >> 5;
    const int mtile = blockIdx.x;                 // 0..624
    const int ntile = blockIdx.y * 4 + wv;        // 0..31
    const int g = lane >> 4;
    const int m = mtile * 16 + (lane & 15);
    const int py = m / IMG_W, px = m % IMG_W;

    v8f acc = {0.f,0.f,0.f,0.f,0.f,0.f,0.f,0.f};
    for (int kb = 0; kb < KB_CONV; ++kb) {
        int K0 = kb * 32;
        int seg = K0 >> 9, coff = K0 & 511;
        int ky = seg / 3, kx = seg % 3;
        const unsigned short* ap =
            xp + ((size_t)(py + ky) * PADW + (px + kx)) * IN_CH + coff + g * 16;
        ABReg a, b;
        a.q[0] = *reinterpret_cast<const u32x4*>(ap);
        a.q[1] = *reinterpret_cast<const u32x4*>(ap + 8);
        const u32* bp = wp + (((size_t)kb * 32 + ntile) * 32 + lane) * 8;
        b.q[0] = *reinterpret_cast<const u32x4*>(bp);
        b.q[1] = *reinterpret_cast<const u32x4*>(bp + 4);
        if (kb + 1 < KB_CONV)
            __builtin_prefetch(wp + (((size_t)(kb + 1) * 32 + ntile) * 32 + lane) * 8, 0, 1);
        acc = __builtin_amdgcn_wmma_f32_16x16x32_bf16(false, a.v, false, b.v,
                                                      (short)0, acc, false, false);
    }
    // D: lane holds N = ntile*16 + lane%16, M = v + 8*g  -> 8 consecutive pixels
    int n = ntile * 16 + (lane & 15);
    float* dst = conv_pre + (size_t)n * NPIX + mtile * 16 + 8 * g;
    f32x4 lo = { acc[0], acc[1], acc[2], acc[3] };
    f32x4 hi = { acc[4], acc[5], acc[6], acc[7] };
    *reinterpret_cast<f32x4*>(dst)     = lo;
    *reinterpret_cast<f32x4*>(dst + 4) = hi;
}

// per-channel batch stats over 10000 pixels
__global__ void k_bn_stats(const float* __restrict__ conv_pre, float* __restrict__ mv) {
    __shared__ float s1[256], s2[256];
    int c = blockIdx.x, tid = threadIdx.x;
    const float* row = conv_pre + (size_t)c * NPIX;
    float sum = 0.f, sq = 0.f;
    for (int i = tid; i < NPIX; i += 256) { float v = row[i]; sum += v; sq += v * v; }
    s1[tid] = sum; s2[tid] = sq; __syncthreads();
    for (int s = 128; s > 0; s >>= 1) {
        if (tid < s) { s1[tid] += s1[tid + s]; s2[tid] += s2[tid + s]; }
        __syncthreads();
    }
    if (tid == 0) {
        float m = s1[0] / (float)NPIX;
        float v = s2[0] / (float)NPIX - m * m;
        mv[c] = m; mv[512 + c] = fmaxf(v, 0.f);
    }
}

// BN + LeakyReLU -> conv1 (fp32 NCHW to d_out) and HWC bf16 activation
__global__ void k_bn_apply(const float* __restrict__ conv_pre, const float* __restrict__ mv,
                           const float* __restrict__ gamma, const float* __restrict__ beta,
                           float* __restrict__ out, unsigned short* __restrict__ c1h) {
    int idx = blockIdx.x * blockDim.x + threadIdx.x;
    if (idx >= MID * NPIX) return;
    int c = idx / NPIX, p = idx % NPIX;
    float m = mv[c], v = mv[512 + c];
    float y = (conv_pre[idx] - m) * rsqrtf(v + BN_EPS) * gamma[c] + beta[c];
    y = (y > 0.f) ? y : 0.1f * y;
    out[OUT_CONV1 + idx] = y;
    c1h[(size_t)p * MID + c] = f2bf(y);
}

// head weights (36 bnd + 18 cls, zero-padded to 64) -> B-operand packed
__global__ void k_pack_wheads(const float* __restrict__ bw, const float* __restrict__ cw,
                              u32* __restrict__ wp) {
    int idx = blockIdx.x * blockDim.x + threadIdx.x;
    if (idx >= KB_HEAD * 4 * 32 * 8) return;
    int d  = idx & 7;
    int l  = (idx >> 3) & 31;
    int nt = (idx >> 8) & 3;
    int kb = idx >> 10;
    int n = nt * 16 + (l & 15);
    int g = l >> 4;
    int K = kb * 32 + g * 16 + 2 * d;
    float v0 = 0.f, v1 = 0.f;
    if (n < 36)      { v0 = bw[(size_t)n * MID + K];        v1 = bw[(size_t)n * MID + K + 1]; }
    else if (n < 54) { v0 = cw[(size_t)(n - 36) * MID + K]; v1 = cw[(size_t)(n - 36) * MID + K + 1]; }
    wp[idx] = (u32)f2bf(v0) | ((u32)f2bf(v1) << 16);
}

// heads GEMM: [10000 x 512] x [512 x 64]
__global__ void __launch_bounds__(128) k_heads_wmma(const unsigned short* __restrict__ c1h,
                                                    const u32* __restrict__ wp,
                                                    float* __restrict__ G) {
    const int lane = threadIdx.x & 31;
    const int ntile = threadIdx.x >> 5;           // 0..3
    const int mtile = blockIdx.x;                 // 0..624
    const int g = lane >> 4;
    const int m = mtile * 16 + (lane & 15);

    v8f acc = {0.f,0.f,0.f,0.f,0.f,0.f,0.f,0.f};
    for (int kb = 0; kb < KB_HEAD; ++kb) {
        const unsigned short* ap = c1h + (size_t)m * MID + kb * 32 + g * 16;
        ABReg a, b;
        a.q[0] = *reinterpret_cast<const u32x4*>(ap);
        a.q[1] = *reinterpret_cast<const u32x4*>(ap + 8);
        const u32* bp = wp + (((size_t)kb * 4 + ntile) * 32 + lane) * 8;
        b.q[0] = *reinterpret_cast<const u32x4*>(bp);
        b.q[1] = *reinterpret_cast<const u32x4*>(bp + 4);
        acc = __builtin_amdgcn_wmma_f32_16x16x32_bf16(false, a.v, false, b.v,
                                                      (short)0, acc, false, false);
    }
    int col = ntile * 16 + (lane & 15);
    for (int v = 0; v < 8; ++v)
        G[(size_t)(mtile * 16 + v + 8 * g) * 64 + col] = acc[v];
}

// softmax, anchor decode, roi clip/validate, histogram
__global__ void k_decode(const float* __restrict__ G,
                         const float* __restrict__ bnd_b, const float* __restrict__ cls_b,
                         float* __restrict__ out, float* __restrict__ rois,
                         float* __restrict__ score, int* __restrict__ hist) {
    int i = blockIdx.x * blockDim.x + threadIdx.x;
    if (i >= NPROP) return;
    int p = i / NANCH, a = i % NANCH;
    const float* gp = G + (size_t)p * 64;
    float t0 = gp[a * 4 + 0] + bnd_b[a * 4 + 0];
    float t1 = gp[a * 4 + 1] + bnd_b[a * 4 + 1];
    float t2 = gp[a * 4 + 2] + bnd_b[a * 4 + 2];
    float t3 = gp[a * 4 + 3] + bnd_b[a * 4 + 3];
    out[OUT_BND + (size_t)i * 4 + 0] = t0;
    out[OUT_BND + (size_t)i * 4 + 1] = t1;
    out[OUT_BND + (size_t)i * 4 + 2] = t2;
    out[OUT_BND + (size_t)i * 4 + 3] = t3;

    float c0 = gp[36 + a * 2 + 0] + cls_b[a * 2 + 0];
    float c1 = gp[36 + a * 2 + 1] + cls_b[a * 2 + 1];
    float mx = fmaxf(c0, c1);
    float e0 = __expf(c0 - mx), e1 = __expf(c1 - mx);
    float inv = 1.0f / (e0 + e1);
    float p1 = e1 * inv;
    out[OUT_CLS + (size_t)i * 2 + 0] = e0 * inv;
    out[OUT_CLS + (size_t)i * 2 + 1] = p1;

    // anchors: generated (W,H,A)-major in reference, combined elementwise
    int q = i / NANCH;
    int d1 = q % IMG_H, d0 = q / IMG_H;
    const float SC[3] = {8.f, 16.f, 32.f};
    const float RT[3] = {0.5f, 1.f, 2.f};
    float s  = SC[a / 3], r = RT[a % 3];
    float sr = sqrtf(r);
    float acy = ((float)d1 + 0.5f) * STRIDE_F;
    float acx = ((float)d0 + 0.5f) * STRIDE_F;
    float ah  = s * sr * STRIDE_F;
    float aw  = s / sr * STRIDE_F;

    float ry = t0 * ah + acy;
    float rx = t1 * aw + acx;
    float rh = __expf(t2) * ah;
    float rw = __expf(t3) * aw;
    float y1 = fminf(fmaxf(ry - rh * 0.5f, 0.f), CLIP_MAX);
    float x1 = fminf(fmaxf(rx - rw * 0.5f, 0.f), CLIP_MAX);
    float y2 = fminf(fmaxf(ry + rh * 0.5f, 0.f), CLIP_MAX);
    float x2 = fminf(fmaxf(rx + rw * 0.5f, 0.f), CLIP_MAX);
    bool valid = ((y2 - y1) >= MIN_SIZE) && ((x2 - x1) >= MIN_SIZE);
    float sc = valid ? p1 : -__builtin_inff();

    rois[(size_t)i * 4 + 0] = y1;
    rois[(size_t)i * 4 + 1] = x1;
    rois[(size_t)i * 4 + 2] = y2;
    rois[(size_t)i * 4 + 3] = x2;
    score[i] = sc;
    atomicAdd(&hist[score_bin(sc)], 1);
}

// find threshold bin for top-12000
__global__ void k_thresh(const int* __restrict__ hist, int* __restrict__ ctrs) {
    if (threadIdx.x != 0) return;
    int cum = 0, T = 0, hi = 0;
    for (int b = NBINS - 1; b >= 0; --b) {
        int c2 = cum + hist[b];
        if (c2 >= PRE_NMS) { T = b; hi = cum; break; }
        cum = c2;
    }
    ctrs[0] = 0; ctrs[1] = 0; ctrs[2] = T; ctrs[3] = hi;
}

// compact top-12000 into selB/selS (order irrelevant for greedy-argmax NMS)
__global__ void k_compact(const float* __restrict__ rois, const float* __restrict__ score,
                          int* __restrict__ ctrs, float* __restrict__ selB,
                          float* __restrict__ selS) {
    int i = blockIdx.x * blockDim.x + threadIdx.x;
    if (i >= NPROP) return;
    int T = ctrs[2], hi = ctrs[3];
    float sc = score[i];
    int b = score_bin(sc);
    int slot = -1;
    if (b > T) slot = atomicAdd(&ctrs[0], 1);
    else if (b == T) {
        int j = atomicAdd(&ctrs[1], 1);
        if (j < PRE_NMS - hi) slot = hi + j;
    }
    if (slot < 0) return;
    selB[(size_t)slot * 4 + 0] = rois[(size_t)i * 4 + 0];
    selB[(size_t)slot * 4 + 1] = rois[(size_t)i * 4 + 1];
    selB[(size_t)slot * 4 + 2] = rois[(size_t)i * 4 + 2];
    selB[(size_t)slot * 4 + 3] = rois[(size_t)i * 4 + 3];
    selS[slot] = sc;
}

// single-block persistent greedy NMS (2000 iterations)
__global__ void __launch_bounds__(1024) k_nms(const float* __restrict__ selB,
                                              float* __restrict__ selS,
                                              float* __restrict__ out) {
    __shared__ float sv[1024];
    __shared__ int   si[1024];
    __shared__ float sb[6];
    const int tid = threadIdx.x;
    const float NEG = -__builtin_inff();

    for (int it = 0; it < POST_NMS; ++it) {
        float bv = NEG; int bi = 0;
        for (int i = tid; i < PRE_NMS; i += 1024) {
            float v = selS[i];
            if (v > bv) { bv = v; bi = i; }
        }
        sv[tid] = bv; si[tid] = bi; __syncthreads();
        for (int s = 512; s > 0; s >>= 1) {
            if (tid < s && sv[tid + s] > sv[tid]) { sv[tid] = sv[tid + s]; si[tid] = si[tid + s]; }
            __syncthreads();
        }
        if (tid == 0) {
            int j = si[0];
            bool ok = sv[0] > NEG;
            float b0 = ok ? selB[(size_t)j * 4 + 0] : 0.f;
            float b1 = ok ? selB[(size_t)j * 4 + 1] : 0.f;
            float b2 = ok ? selB[(size_t)j * 4 + 2] : 0.f;
            float b3 = ok ? selB[(size_t)j * 4 + 3] : 0.f;
            out[OUT_ROI + (size_t)it * 4 + 0] = b0;
            out[OUT_ROI + (size_t)it * 4 + 1] = b1;
            out[OUT_ROI + (size_t)it * 4 + 2] = b2;
            out[OUT_ROI + (size_t)it * 4 + 3] = b3;
            sb[0] = b0; sb[1] = b1; sb[2] = b2; sb[3] = b3;
            sb[4] = (b2 - b0) * (b3 - b1);
            sb[5] = ok ? 1.f : 0.f;
            if (ok) selS[j] = NEG;
        }
        __syncthreads();
        if (sb[5] > 0.f) {
            float by1 = sb[0], bx1 = sb[1], by2 = sb[2], bx2 = sb[3], ba = sb[4];
            for (int i = tid; i < PRE_NMS; i += 1024) {
                if (selS[i] > NEG) {
                    float y1 = selB[(size_t)i * 4 + 0];
                    float x1 = selB[(size_t)i * 4 + 1];
                    float y2 = selB[(size_t)i * 4 + 2];
                    float x2 = selB[(size_t)i * 4 + 3];
                    float iy = fminf(by2, y2) - fmaxf(by1, y1);
                    float ix = fminf(bx2, x2) - fmaxf(bx1, x1);
                    float inter = fmaxf(iy, 0.f) * fmaxf(ix, 0.f);
                    float area = (y2 - y1) * (x2 - x1);
                    float iou = inter / (area + ba - inter + 1e-9f);
                    if (iou > NMS_THR) selS[i] = NEG;
                }
            }
        }
        __syncthreads();
    }
}

// ---------- launcher ----------
extern "C" void kernel_launch(void* const* d_in, const int* in_sizes, int n_in,
                              void* d_out, int out_size, void* d_ws, size_t ws_size,
                              hipStream_t stream) {
    (void)in_sizes; (void)n_in; (void)out_size; (void)ws_size;
    const float* x      = (const float*)d_in[0];
    const float* conv_w = (const float*)d_in[1];
    const float* gamma  = (const float*)d_in[2];
    const float* beta   = (const float*)d_in[3];
    const float* bnd_w  = (const float*)d_in[4];
    const float* bnd_b  = (const float*)d_in[5];
    const float* cls_w  = (const float*)d_in[6];
    const float* cls_b  = (const float*)d_in[7];
    float* out = (float*)d_out;

    char* ws = (char*)d_ws;
    unsigned short* xp   = (unsigned short*)(ws + XP_OFF);
    u32*   wpc   = (u32*)  (ws + WC_OFF);
    float* cpre  = (float*)(ws + CP_OFF);
    float* mv    = (float*)(ws + MV_OFF);
    unsigned short* c1h  = (unsigned short*)(ws + C1H_OFF);
    u32*   wph   = (u32*)  (ws + WH_OFF);
    float* G     = (float*)(ws + G_OFF);
    float* rois  = (float*)(ws + ROIS_OFF);
    float* score = (float*)(ws + SCORE_OFF);
    int*   hist  = (int*)  (ws + HIST_OFF);
    int*   ctrs  = (int*)  (ws + CTRS_OFF);
    float* selB  = (float*)(ws + SELB_OFF);
    float* selS  = (float*)(ws + SELS_OFF);

    k_init<<<(NBINS + 16 + 255) / 256, 256, 0, stream>>>(hist);  // hist+ctrs contiguous
    k_pack_x<<<(PADW * PADW * IN_CH) / 256, 256, 0, stream>>>(x, xp);
    k_pack_wconv<<<(KB_CONV * 32 * 32 * 8) / 256, 256, 0, stream>>>(conv_w, wpc);
    k_conv_wmma<<<dim3(625, 8), 128, 0, stream>>>(xp, wpc, cpre);
    k_bn_stats<<<512, 256, 0, stream>>>(cpre, mv);
    k_bn_apply<<<(MID * NPIX) / 256, 256, 0, stream>>>(cpre, mv, gamma, beta, out, c1h);
    k_pack_wheads<<<(KB_HEAD * 4 * 32 * 8) / 256, 256, 0, stream>>>(bnd_w, cls_w, wph);
    k_heads_wmma<<<625, 128, 0, stream>>>(c1h, wph, G);
    k_decode<<<(NPROP + 255) / 256, 256, 0, stream>>>(G, bnd_b, cls_b, out, rois, score, hist);
    k_thresh<<<1, 32, 0, stream>>>(hist, ctrs);
    k_compact<<<(NPROP + 255) / 256, 256, 0, stream>>>(rois, score, ctrs, selB, selS);
    k_nms<<<1, 1024, 0, stream>>>(selB, selS, out);
}
